// LoR_QKVTranslution_4724464025979
// MI455X (gfx1250) — matrix operand order repaired and checked by reading, not compiled, not run on hardware
//
#include <hip/hip_runtime.h>
#include <hip/hip_bf16.h>

#define SEQ 512
#define HEADS 8
#define DH 64
#define DR 16
#define HD 128     // HEADS*DR
#define INNER 512  // HEADS*DH

typedef __bf16 bf16_t;
typedef __attribute__((ext_vector_type(16))) __bf16 bf16x16;
typedef __attribute__((ext_vector_type(8)))  __bf16 bf16x8;
typedef __attribute__((ext_vector_type(8)))  float  f32x8;

// ---------- bf16 helpers (hardware cvt) ----------
__device__ __forceinline__ bf16_t f2bf(float f) { return (bf16_t)f; }
__device__ __forceinline__ float bf2f(bf16_t b) {
  unsigned short h = __builtin_bit_cast(unsigned short, b);
  unsigned u = ((unsigned)h) << 16;
  return __builtin_bit_cast(float, u);
}

__device__ __forceinline__ f32x8 wmma_bf16(bf16x16 a, bf16x16 b, f32x8 c) {
  return __builtin_amdgcn_wmma_f32_16x16x32_bf16(false, a, false, b, (short)0, c,
                                                 false, false);
}

// ---------- per-lane tile loaders (CDNA5 WMMA register layouts) ----------
// A (16x32 bf16): lane L: row=L&15, half=L>>4; a[i]=A[row][half*8+i], a[8+i]=A[row][16+half*8+i]
__device__ __forceinline__ bf16x16 load_a_bf16(const bf16_t* A, int lda, int row0,
                                               int row_max, int k) {
  int lane = threadIdx.x & 31;
  int row = row0 + (lane & 15);
  if (row > row_max) row = row_max;
  int half = lane >> 4;
  const bf16_t* p = A + (size_t)row * lda + k + half * 8;
  bf16x8 lo = *reinterpret_cast<const bf16x8*>(p);
  bf16x8 hi = *reinterpret_cast<const bf16x8*>(p + 16);
  bf16x16 a;
#pragma unroll
  for (int i = 0; i < 8; i++) { a[i] = lo[i]; a[8 + i] = hi[i]; }
  return a;
}

__device__ __forceinline__ bf16x16 load_a_f32(const float* A, int lda, int row0, int k) {
  int lane = threadIdx.x & 31;
  int row = row0 + (lane & 15);
  int half = lane >> 4;
  const float* p = A + (size_t)row * lda + k + half * 8;
  bf16x16 a;
#pragma unroll
  for (int i = 0; i < 8; i++) { a[i] = f2bf(p[i]); a[8 + i] = f2bf(p[16 + i]); }
  return a;
}

// B (32x16): lane L: col=L&15, half=L>>4; b[i]=B[half*8+i][col], b[8+i]=B[16+half*8+i][col].
// Source S is N-major: B[k][n] = S[n][k] (contiguous along k) -> two 16B loads.
__device__ __forceinline__ bf16x16 load_b_t(const bf16_t* S, int lds_, int n0, int k) {
  int lane = threadIdx.x & 31;
  int col = n0 + (lane & 15);
  int half = lane >> 4;
  const bf16_t* p = S + (size_t)col * lds_ + k + half * 8;
  bf16x8 lo = *reinterpret_cast<const bf16x8*>(p);
  bf16x8 hi = *reinterpret_cast<const bf16x8*>(p + 16);
  bf16x16 b;
#pragma unroll
  for (int i = 0; i < 8; i++) { b[i] = lo[i]; b[8 + i] = hi[i]; }
  return b;
}

// C/D (16x16 f32): lane L: col=L&15, half=L>>4; acc[v] -> row=half*8+v
__device__ __forceinline__ void store_c_bf(bf16_t* C, int ldc, int row0, int col0,
                                           f32x8 acc) {
  int lane = threadIdx.x & 31;
  int col = col0 + (lane & 15);
  int half = lane >> 4;
#pragma unroll
  for (int v = 0; v < 8; v++)
    C[(size_t)(row0 + half * 8 + v) * ldc + col] = f2bf(acc[v]);
}

__device__ __forceinline__ void store_c_bf_masked(bf16_t* C, size_t row_stride, int row0,
                                                  int row_lim, size_t col_off, f32x8 acc) {
  int lane = threadIdx.x & 31;
  int col = lane & 15;
  int half = lane >> 4;
#pragma unroll
  for (int v = 0; v < 8; v++) {
    int row = row0 + half * 8 + v;
    if (row < row_lim) C[(size_t)row * row_stride + col_off + col] = f2bf(acc[v]);
  }
}

// ---------- K0a: transpose + f32->bf16: WT[n][k] = W[k][n] ----------
__global__ void tcvt_kernel(const float* __restrict__ W, bf16_t* __restrict__ WT,
                            int K, int N) {
  __shared__ bf16_t t[16][17];
  int nt = blockIdx.x * 16, kt = blockIdx.y * 16;
  const float* Wb = W + (size_t)blockIdx.z * K * N;
  bf16_t* WTb = WT + (size_t)blockIdx.z * K * N;
  t[threadIdx.y][threadIdx.x] =
      f2bf(Wb[(size_t)(kt + threadIdx.y) * N + nt + threadIdx.x]);
  __syncthreads();
  WTb[(size_t)(nt + threadIdx.y) * K + kt + threadIdx.x] = t[threadIdx.x][threadIdx.y];
}

// ---------- K0b: bf16 transpose: D[c][r] = S[r][c] ----------
__global__ void tbf_kernel(const bf16_t* __restrict__ S, int ldS,
                           bf16_t* __restrict__ D, int ldD) {
  __shared__ bf16_t t[16][17];
  int ct = blockIdx.x * 16, rt = blockIdx.y * 16;
  t[threadIdx.y][threadIdx.x] = S[(size_t)(rt + threadIdx.y) * ldS + ct + threadIdx.x];
  __syncthreads();
  D[(size_t)(ct + threadIdx.y) * ldD + rt + threadIdx.x] = t[threadIdx.x][threadIdx.y];
}

// ---------- K1: LayerNorm -> bf16 ----------
__global__ void ln_kernel(const float* __restrict__ x, const float* __restrict__ w,
                          const float* __restrict__ b, bf16_t* __restrict__ xn) {
  __shared__ float red[256];
  int n = blockIdx.x;
  const float* xr = x + (size_t)n * 512;
  float s = 0.f;
  for (int i = threadIdx.x; i < 512; i += 256) s += xr[i];
  red[threadIdx.x] = s; __syncthreads();
  for (int off = 128; off > 0; off >>= 1) {
    if (threadIdx.x < off) red[threadIdx.x] += red[threadIdx.x + off];
    __syncthreads();
  }
  float mu = red[0] * (1.f / 512.f);
  __syncthreads();
  float v = 0.f;
  for (int i = threadIdx.x; i < 512; i += 256) { float d = xr[i] - mu; v += d * d; }
  red[threadIdx.x] = v; __syncthreads();
  for (int off = 128; off > 0; off >>= 1) {
    if (threadIdx.x < off) red[threadIdx.x] += red[threadIdx.x + off];
    __syncthreads();
  }
  float rstd = rsqrtf(red[0] * (1.f / 512.f) + 1e-5f);
  for (int i = threadIdx.x; i < 512; i += 256)
    xn[(size_t)n * 512 + i] = f2bf((xr[i] - mu) * rstd * w[i] + b[i]);
}

// ---------- K2: C_bf16 = A_bf16(Mx512) @ WT_bf16(N-major), 2M x 4N tiles ----------
__global__ void gemm_aw_kernel(const bf16_t* __restrict__ A, int lda,
                               const bf16_t* __restrict__ WT,
                               bf16_t* __restrict__ C, int ldc) {
  int nt = blockIdx.x * 64, mt = blockIdx.y * 32;
  f32x8 acc[2][4] = {};
  for (int k = 0; k < 512; k += 32) {
    bf16x16 a0 = load_a_bf16(A, lda, mt, 511, k);
    bf16x16 a1 = load_a_bf16(A, lda, mt + 16, 511, k);
    bf16x16 b[4];
#pragma unroll
    for (int j = 0; j < 4; j++) b[j] = load_b_t(WT, 512, nt + j * 16, k);
#pragma unroll
    for (int j = 0; j < 4; j++) {
      acc[0][j] = wmma_bf16(a0, b[j], acc[0][j]);
      acc[1][j] = wmma_bf16(a1, b[j], acc[1][j]);
    }
  }
#pragma unroll
  for (int j = 0; j < 4; j++) {
    store_c_bf(C, ldc, mt, nt + j * 16, acc[0][j]);
    store_c_bf(C, ldc, mt + 16, nt + j * 16, acc[1][j]);
  }
}

// ---------- K3: batched rel GEMMs: G[row][r][:] = a[row] @ P[sel(r)] ----------
// full 16x128 strip per wave; loads split from WMMAs for overlap
__global__ void rel_u_kernel(const bf16_t* __restrict__ A, const bf16_t* __restrict__ PT,
                             bf16_t* __restrict__ G, int qmode) {
  int r = blockIdx.y;
  int row0, row1, pidx;
  if (qmode) { row0 = r; row1 = 512; pidx = 511 - r; }
  else       { row0 = 0; row1 = 512 - r; pidx = r; }
  int mt = row0 + 16 * blockIdx.x;
  if (mt >= row1) return;
  const bf16_t* Pt = PT + (size_t)pidx * HD * HD;  // [e][f] bf16
  f32x8 acc[8] = {};
#pragma unroll
  for (int k = 0; k < HD; k += 32) {
    bf16x16 a = load_a_bf16(A, HD, mt, 511, k);
    bf16x16 b[8];
#pragma unroll
    for (int j = 0; j < 8; j++) b[j] = load_b_t(Pt, HD, j * 16, k);
#pragma unroll
    for (int j = 0; j < 8; j++) acc[j] = wmma_bf16(a, b[j], acc[j]);
  }
  bf16_t* Gr = G + (size_t)r * HD;  // row n at stride 512*HD
#pragma unroll
  for (int j = 0; j < 8; j++)
    store_c_bf_masked(Gr, (size_t)512 * HD, mt, row1, j * 16, acc[j]);
}

// ---------- K4: logits = (q1k1 + rel-dot)/2 * dh^-0.5, causal mask ----------
__global__ void dots_kernel(const bf16_t* __restrict__ qkv, const bf16_t* __restrict__ Gq,
                            const bf16_t* __restrict__ Gk, float* __restrict__ dots) {
  int h = blockIdx.z;
  int nt = blockIdx.y * 16, mt = blockIdx.x * 16;
  float* drow = dots + ((size_t)h * 512) * 512;
  int lane = threadIdx.x & 31;
  int col = lane & 15, half = lane >> 4;
  if (mt > nt + 15) {  // fully above diagonal: all -inf
#pragma unroll
    for (int v = 0; v < 8; v++)
      drow[(size_t)(nt + half * 8 + v) * 512 + mt + col] = -__builtin_inff();
    return;
  }
  bf16x16 a[2], b[2];
#pragma unroll
  for (int s = 0; s < 2; s++) {
    a[s] = load_a_bf16(qkv + h * DH, 3 * INNER, nt, 511, s * 32);          // q1
    b[s] = load_b_t(qkv + INNER + h * DH, 3 * INNER, mt, s * 32);          // k1^T
  }
  f32x8 acc = {};
#pragma unroll
  for (int s = 0; s < 2; s++) acc = wmma_bf16(a[s], b[s], acc);
  const float scale = 0.0625f;  // 0.5 * 64^-0.5
#pragma unroll
  for (int v = 0; v < 8; v++) {
    int n = nt + half * 8 + v, m = mt + col;
    float d;
    if (m <= n) {
      int r = n - m;
      const bf16_t* gq = Gq + ((size_t)n * 512 + r) * HD + h * DR;
      const bf16_t* gk = Gk + ((size_t)m * 512 + r) * HD + h * DR;
      float d2 = 0.f;
#pragma unroll
      for (int e = 0; e < DR; e++) d2 += bf2f(gq[e]) * bf2f(gk[e]);
      d = (acc[v] + d2) * scale;
    } else {
      d = -__builtin_inff();
    }
    drow[(size_t)n * 512 + m] = d;
  }
}

// ---------- K5: row softmax -> bf16 ----------
__global__ void softmax_kernel(const float* __restrict__ dots, bf16_t* __restrict__ attn) {
  __shared__ float red[256];
  __shared__ float ev[512];
  int n = blockIdx.x, h = blockIdx.y;
  const float* row = dots + ((size_t)h * 512 + n) * 512;
  float mx = -__builtin_inff();
  for (int i = threadIdx.x; i < 512; i += 256) mx = fmaxf(mx, row[i]);
  red[threadIdx.x] = mx; __syncthreads();
  for (int off = 128; off > 0; off >>= 1) {
    if (threadIdx.x < off) red[threadIdx.x] = fmaxf(red[threadIdx.x], red[threadIdx.x + off]);
    __syncthreads();
  }
  mx = red[0]; __syncthreads();
  float s = 0.f;
  for (int i = threadIdx.x; i < 512; i += 256) {
    float e = __expf(row[i] - mx);
    ev[i] = e;
    s += e;
  }
  red[threadIdx.x] = s; __syncthreads();
  for (int off = 128; off > 0; off >>= 1) {
    if (threadIdx.x < off) red[threadIdx.x] += red[threadIdx.x + off];
    __syncthreads();
  }
  float inv = 1.f / red[0];
  bf16_t* arow = attn + ((size_t)h * 512 + n) * 512;
  for (int i = threadIdx.x; i < 512; i += 256) arow[i] = f2bf(ev[i] * inv);
}

// ---------- K6: out1 = attn @ v1 (per head), 16x64 strip per wave ----------
__global__ void out1_kernel(const bf16_t* __restrict__ attn, const bf16_t* __restrict__ V1T,
                            float* __restrict__ O1) {
  int h = blockIdx.y;
  int nt = blockIdx.x * 16;
  const bf16_t* A = attn + (size_t)h * 512 * 512;
  const bf16_t* Vt = V1T + (size_t)h * DH * 512;  // [d][m] for this head
  f32x8 acc[4] = {};
  for (int k = 0; k < nt + 16; k += 32) {  // causal: attn zero beyond n
    bf16x16 a = load_a_bf16(A, 512, nt, 511, k);
    bf16x16 b[4];
#pragma unroll
    for (int j = 0; j < 4; j++) b[j] = load_b_t(Vt, 512, j * 16, k);
#pragma unroll
    for (int j = 0; j < 4; j++) acc[j] = wmma_bf16(a, b[j], acc[j]);
  }
  int lane = threadIdx.x & 31;
  int half = lane >> 4;
#pragma unroll
  for (int j = 0; j < 4; j++) {
    int col = h * DH + j * 16 + (lane & 15);
#pragma unroll
    for (int v = 0; v < 8; v++)
      O1[(size_t)(nt + half * 8 + v) * INNER + col] = acc[j][v];
  }
}

// ---------- K7: out2 heads = gathered attn . Gv, f32 [n][128] ----------
__global__ void out2_kernel(const bf16_t* __restrict__ attn, const bf16_t* __restrict__ Gv,
                            float* __restrict__ O2H) {
  int n = blockIdx.x;
  int h = threadIdx.x >> 4, e = threadIdx.x & 15;
  const bf16_t* arow = attn + ((size_t)h * 512 + n) * 512;
  float acc = 0.f;
  for (int m = 0; m <= n; m++) {
    float a = bf2f(arow[m]);
    float g = bf2f(Gv[((size_t)m * 512 + (n - m)) * HD + h * DR + e]);
    acc += a * g;
  }
  O2H[(size_t)n * HD + h * DR + e] = acc;
}

// ---------- K8: OC = 0.5*(O1 + O2H @ w_v3) -> bf16, 4 N-tiles ----------
__global__ void combine_kernel(const float* __restrict__ O2H, const bf16_t* __restrict__ WTv3,
                               const float* __restrict__ O1, bf16_t* __restrict__ OC) {
  int jt = blockIdx.x * 64, nt = blockIdx.y * 16;
  f32x8 acc[4] = {};
#pragma unroll
  for (int k = 0; k < HD; k += 32) {
    bf16x16 a = load_a_f32(O2H, HD, nt, k);
    bf16x16 b[4];
#pragma unroll
    for (int j = 0; j < 4; j++) b[j] = load_b_t(WTv3, HD, jt + j * 16, k);
#pragma unroll
    for (int j = 0; j < 4; j++) acc[j] = wmma_bf16(a, b[j], acc[j]);
  }
  int lane = threadIdx.x & 31;
  int half = lane >> 4;
#pragma unroll
  for (int j = 0; j < 4; j++) {
    int col = jt + j * 16 + (lane & 15);
#pragma unroll
    for (int v = 0; v < 8; v++) {
      int row = nt + half * 8 + v;
      OC[(size_t)row * INNER + col] =
          f2bf(0.5f * (O1[(size_t)row * INNER + col] + acc[j][v]));
    }
  }
}

// ---------- K9: out = OC @ w_out + b_out (f32), 2M x 4N tiles ----------
__global__ void final_kernel(const bf16_t* __restrict__ OC, const bf16_t* __restrict__ WTout,
                             const float* __restrict__ b_out, float* __restrict__ out) {
  int dt = blockIdx.x * 64, nt = blockIdx.y * 32;
  f32x8 acc[2][4] = {};
  for (int k = 0; k < 512; k += 32) {
    bf16x16 a0 = load_a_bf16(OC, 512, nt, 511, k);
    bf16x16 a1 = load_a_bf16(OC, 512, nt + 16, 511, k);
    bf16x16 b[4];
#pragma unroll
    for (int j = 0; j < 4; j++) b[j] = load_b_t(WTout, 512, dt + j * 16, k);
#pragma unroll
    for (int j = 0; j < 4; j++) {
      acc[0][j] = wmma_bf16(a0, b[j], acc[0][j]);
      acc[1][j] = wmma_bf16(a1, b[j], acc[1][j]);
    }
  }
  int lane = threadIdx.x & 31;
  int half = lane >> 4;
#pragma unroll
  for (int j = 0; j < 4; j++) {
    int col = dt + j * 16 + (lane & 15);
    float bias = b_out[col];
#pragma unroll
    for (int v = 0; v < 8; v++) {
      out[(size_t)(nt + half * 8 + v) * 512 + col] = acc[0][j][v] + bias;
      out[(size_t)(nt + 16 + half * 8 + v) * 512 + col] = acc[1][j][v] + bias;
    }
  }
}

// ---------- workspace layout ----------
static constexpr size_t MiB = 1ull << 20;
static constexpr size_t OFF_XN    = 0;                       // 512*512 bf16 (0.5 MiB)
static constexpr size_t OFF_QKV   = 1 * MiB;                 // 512*1536 bf16 (1.5 MiB)
static constexpr size_t OFF_AQ    = 3 * MiB;                 // 512*128 bf16 ×3
static constexpr size_t OFF_AK    = OFF_AQ + 512ull * HD * 2;
static constexpr size_t OFF_AV    = OFF_AK + 512ull * HD * 2;
static constexpr size_t OFF_WTQKV = 4 * MiB;                 // 1536*512 bf16 (1.5 MiB)
static constexpr size_t OFF_WTQ1  = 6 * MiB;                 // 128*512 bf16 ×3
static constexpr size_t OFF_WTK1  = OFF_WTQ1 + 128ull * 512 * 2;
static constexpr size_t OFF_WTV1  = OFF_WTK1 + 128ull * 512 * 2;
static constexpr size_t OFF_WTV3  = 7 * MiB;                 // 512*128 bf16
static constexpr size_t OFF_WTOUT = OFF_WTV3 + 512ull * 128 * 2;  // 512*512 bf16
static constexpr size_t OFF_PTQ   = 8 * MiB;                 // 512*128*128 bf16 (16 MiB) ×3
static constexpr size_t PTSIZE    = 512ull * HD * HD * 2;
static constexpr size_t OFF_PTK   = OFF_PTQ + PTSIZE;
static constexpr size_t OFF_PTV   = OFF_PTK + PTSIZE;
static constexpr size_t GSIZE     = 512ull * 512 * HD * 2;   // 64 MiB each
static constexpr size_t OFF_GQ    = 56 * MiB;
static constexpr size_t OFF_GK    = OFF_GQ + GSIZE;
static constexpr size_t OFF_GV    = OFF_GK + GSIZE;
static constexpr size_t OFF_DOTS  = OFF_GV + GSIZE;          // 8*512*512 f32 (8 MiB)
static constexpr size_t OFF_ATTN  = OFF_DOTS + 8ull * 512 * 512 * 4;  // bf16 (4 MiB)
static constexpr size_t OFF_O1    = OFF_ATTN + 8ull * 512 * 512 * 2;  // f32 (1 MiB)
static constexpr size_t OFF_O2H   = OFF_O1 + 512ull * 512 * 4;
static constexpr size_t OFF_OC    = OFF_O2H + 512ull * HD * 4;
static constexpr size_t OFF_V1T   = OFF_OC + 512ull * 512 * 2;  // v1^T bf16 (0.5 MiB)

extern "C" void kernel_launch(void* const* d_in, const int* in_sizes, int n_in,
                              void* d_out, int out_size, void* d_ws, size_t ws_size,
                              hipStream_t stream) {
  const float* x     = (const float*)d_in[0];
  const float* ln_w  = (const float*)d_in[1];
  const float* ln_b  = (const float*)d_in[2];
  const float* w_qkv = (const float*)d_in[3];
  const float* w_q1  = (const float*)d_in[4];
  const float* p_q2  = (const float*)d_in[5];
  const float* w_k1  = (const float*)d_in[6];
  const float* p_k2  = (const float*)d_in[7];
  const float* w_v1  = (const float*)d_in[8];
  const float* p_v2  = (const float*)d_in[9];
  const float* w_v3  = (const float*)d_in[10];
  const float* w_out = (const float*)d_in[11];
  const float* b_out = (const float*)d_in[12];
  float* out = (float*)d_out;
  char* ws = (char*)d_ws;

  bf16_t* XN    = (bf16_t*)(ws + OFF_XN);
  bf16_t* QKV   = (bf16_t*)(ws + OFF_QKV);
  bf16_t* AQ    = (bf16_t*)(ws + OFF_AQ);
  bf16_t* AK    = (bf16_t*)(ws + OFF_AK);
  bf16_t* AV    = (bf16_t*)(ws + OFF_AV);
  bf16_t* WTQKV = (bf16_t*)(ws + OFF_WTQKV);
  bf16_t* WTQ1  = (bf16_t*)(ws + OFF_WTQ1);
  bf16_t* WTK1  = (bf16_t*)(ws + OFF_WTK1);
  bf16_t* WTV1  = (bf16_t*)(ws + OFF_WTV1);
  bf16_t* WTV3  = (bf16_t*)(ws + OFF_WTV3);
  bf16_t* WTOUT = (bf16_t*)(ws + OFF_WTOUT);
  bf16_t* PTQ   = (bf16_t*)(ws + OFF_PTQ);
  bf16_t* PTK   = (bf16_t*)(ws + OFF_PTK);
  bf16_t* PTV   = (bf16_t*)(ws + OFF_PTV);
  bf16_t* GQ    = (bf16_t*)(ws + OFF_GQ);
  bf16_t* GK    = (bf16_t*)(ws + OFF_GK);
  bf16_t* GV    = (bf16_t*)(ws + OFF_GV);
  float*  DOTS  = (float*)(ws + OFF_DOTS);
  bf16_t* ATTN  = (bf16_t*)(ws + OFF_ATTN);
  float*  O1    = (float*)(ws + OFF_O1);
  float*  O2H   = (float*)(ws + OFF_O2H);
  bf16_t* OC    = (bf16_t*)(ws + OFF_OC);
  bf16_t* V1T   = (bf16_t*)(ws + OFF_V1T);

  dim3 tb(16, 16);
  // weights -> N-major bf16
  tcvt_kernel<<<dim3(96, 32, 1), tb, 0, stream>>>(w_qkv, WTQKV, 512, 1536);
  tcvt_kernel<<<dim3(8, 32, 1), tb, 0, stream>>>(w_q1, WTQ1, 512, HD);
  tcvt_kernel<<<dim3(8, 32, 1), tb, 0, stream>>>(w_k1, WTK1, 512, HD);
  tcvt_kernel<<<dim3(8, 32, 1), tb, 0, stream>>>(w_v1, WTV1, 512, HD);
  tcvt_kernel<<<dim3(32, 8, 1), tb, 0, stream>>>(w_v3, WTV3, HD, INNER);
  tcvt_kernel<<<dim3(32, 32, 1), tb, 0, stream>>>(w_out, WTOUT, 512, 512);
  // P tensors -> per-l [e][f] bf16
  tcvt_kernel<<<dim3(8, 8, 512), tb, 0, stream>>>(p_q2, PTQ, HD, HD);
  tcvt_kernel<<<dim3(8, 8, 512), tb, 0, stream>>>(p_k2, PTK, HD, HD);
  tcvt_kernel<<<dim3(8, 8, 512), tb, 0, stream>>>(p_v2, PTV, HD, HD);

  ln_kernel<<<512, 256, 0, stream>>>(x, ln_w, ln_b, XN);

  gemm_aw_kernel<<<dim3(24, 16), 32, 0, stream>>>(XN, 512, WTQKV, QKV, 1536);
  gemm_aw_kernel<<<dim3(2, 16), 32, 0, stream>>>(XN, 512, WTQ1, AQ, HD);
  gemm_aw_kernel<<<dim3(2, 16), 32, 0, stream>>>(XN, 512, WTK1, AK, HD);
  gemm_aw_kernel<<<dim3(2, 16), 32, 0, stream>>>(XN, 512, WTV1, AV, HD);

  // v1 (rows m, cols h*64+d within QKV) -> V1T[d][m]
  tbf_kernel<<<dim3(32, 32), tb, 0, stream>>>(QKV + 2 * INNER, 3 * INNER, V1T, 512);

  rel_u_kernel<<<dim3(32, 512), 32, 0, stream>>>(AQ, PTQ, GQ, 1);
  rel_u_kernel<<<dim3(32, 512), 32, 0, stream>>>(AK, PTK, GK, 0);
  rel_u_kernel<<<dim3(32, 512), 32, 0, stream>>>(AV, PTV, GV, 0);

  dots_kernel<<<dim3(32, 32, 8), 32, 0, stream>>>(QKV, GQ, GK, DOTS);
  softmax_kernel<<<dim3(512, 8), 256, 0, stream>>>(DOTS, ATTN);

  out1_kernel<<<dim3(32, 8), 32, 0, stream>>>(ATTN, V1T, O1);
  out2_kernel<<<512, 128, 0, stream>>>(ATTN, GV, O2H);

  combine_kernel<<<dim3(8, 32), 32, 0, stream>>>(O2H, WTV3, O1, OC);
  final_kernel<<<dim3(8, 16), 32, 0, stream>>>(OC, WTOUT, b_out, out);
  (void)in_sizes; (void)n_in; (void)out_size; (void)ws_size;
}
